// BILSTMCRF_31464930411203
// MI455X (gfx1250) — matrix operand order repaired and checked by reading, not compiled, not run on hardware
//
#include <hip/hip_runtime.h>

// ---- problem constants (match reference) ----
#define V_   32
#define E_   128
#define H_   128
#define N_   8
#define B_   64
#define T_   1024
#define G4H  512   // 4*H

typedef __attribute__((ext_vector_type(16))) _Float16 v16h;
typedef __attribute__((ext_vector_type(8)))  _Float16 v8h;
typedef __attribute__((ext_vector_type(8)))  float    v8f;
typedef __attribute__((ext_vector_type(4)))  float    f32x4;

struct Frag32 { f32x4 a, b; };

// Build a 16-element f16 WMMA fragment from two 16-byte chunks.
static __device__ __forceinline__ v16h ldfrag(const _Float16* p0, const _Float16* p1) {
  Frag32 t;
  t.a = *(const f32x4*)p0;
  t.b = *(const f32x4*)p1;
  return __builtin_bit_cast(v16h, t);
}

static __device__ __forceinline__ float sigmoidf_(float x) {
  return 1.0f / (1.0f + __expf(-x));
}

#define WMMA_F16(a, b, c) \
  __builtin_amdgcn_wmma_f32_16x16x32_f16(false, (a), false, (b), (short)0, (c), false, false)

// x_proj stored pre-permuted into C-fragment layout:
//   tile index ti = ((t*4 + m)*4 + G)*8 + q8   (m=M-tile, G=gate, q8=16-col group)
//   element off  = ti*256 + lane*8             (8 contiguous f16 per lane = one b128)
static __device__ __forceinline__ size_t xp_off(int t, int m, int G, int q8, int lane) {
  return (((size_t)((t * 4 + m) * 4 + G) * 8 + q8) * 256) + lane * 8;
}

// ---------------------------------------------------------------------------
// f32 -> f16 weight / embedding conversion
// ---------------------------------------------------------------------------
__global__ void k_f32_to_f16(const float* __restrict__ src, _Float16* __restrict__ dst, int n) {
  int i = blockIdx.x * blockDim.x + threadIdx.x;
  if (i < n) dst[i] = (_Float16)src[i];
}

// ---------------------------------------------------------------------------
// Embedding gather + input projection:  xproj[t,b,:] = W_ih @ emb[x[b,t]] + bias
// One block per timestep t (64 rows = full batch), blockIdx.y = direction.
// 16 waves: wave = (wm, wn); wave computes M-tile wm (16 batch rows) x 8 N-tiles.
// Output written in permuted C-fragment layout (16B per tile per lane).
// ---------------------------------------------------------------------------
__global__ void __launch_bounds__(512)
k_input_proj(const int* __restrict__ x, const _Float16* __restrict__ emb16,
             const _Float16* __restrict__ wih16_f, const _Float16* __restrict__ wih16_b,
             const float* __restrict__ bias_f, const float* __restrict__ bias_b,
             _Float16* __restrict__ xproj_f, _Float16* __restrict__ xproj_b)
{
  const int t   = blockIdx.x;
  const int dir = blockIdx.y;
  const _Float16* wih16 = dir ? wih16_b : wih16_f;
  const float*    bias  = dir ? bias_b  : bias_f;
  _Float16*       xproj = dir ? xproj_b : xproj_f;

  const int tid  = threadIdx.x;
  const int w    = tid >> 5, lane = tid & 31;
  const int wm   = w >> 2,   wn   = w & 3;
  const int ln   = lane & 15, half = lane >> 4;

  // A-matrix row for this lane = batch index; gather embedding row of its token
  const int brow_a = wm * 16 + ln;
  const int token  = x[brow_a * T_ + t];
  const _Float16* embrow = emb16 + token * E_;

  v8f acc[8];
  #pragma unroll
  for (int j = 0; j < 8; ++j)
    #pragma unroll
    for (int v = 0; v < 8; ++v) acc[j][v] = 0.0f;

  #pragma unroll
  for (int k = 0; k < 4; ++k) {                       // K = 128 in 4 x 32 steps
    // 16-bit A 16x32 layout: lane holds K = half*8 + [0..7] and 16+half*8 + [0..7]
    const _Float16* ap = embrow + k * 32 + half * 8;
    v16h afrag = ldfrag(ap, ap + 16);
    #pragma unroll
    for (int j = 0; j < 8; ++j) {
      // 16-bit B 32x16 layout: lane = column, holds K = half*8 + [0..7] and 16+half*8 + [0..7]
      const int n = wn * 128 + j * 16 + ln;           // W_ih row == output column
      const _Float16* wp = wih16 + (size_t)n * E_ + k * 32 + half * 8;
      v16h bfrag = ldfrag(wp, wp + 16);
      acc[j] = WMMA_F16(afrag, bfrag, acc[j]);
    }
  }

  // Store C fragments in permuted layout: one contiguous v8h per tile per lane.
  #pragma unroll
  for (int j = 0; j < 8; ++j) {
    const int col  = wn * 128 + j * 16 + ln;
    const float bc = bias[col];
    v8h hv;
    #pragma unroll
    for (int v = 0; v < 8; ++v) hv[v] = (_Float16)(acc[j][v] + bc);
    *(v8h*)(xproj + xp_off(t, wm, /*G=*/wn, /*q8=*/j, lane)) = hv;
  }
}

// ---------------------------------------------------------------------------
// Persistent LSTM recurrence. One block per direction, 512 threads = 16 waves.
// Wave (mp = w>>3, q8 = w&7) owns TWO M-tiles (2*mp, 2*mp+1) and the SAME
// 16 hidden columns in each of the 4 gates => 8 output tiles, and only
// 4 gates x 4 k = 16 B-fragments, which are loaded ONCE and stay
// register-resident for all 1024 timesteps (no per-step W traffic at all).
// Per-step LDS traffic is just the h A-fragments (16 x b128 per wave).
// Cell update stays fully in-register (gate tiles are lane-aligned).
// ---------------------------------------------------------------------------
__global__ void __launch_bounds__(512)
k_lstm_recur(const _Float16* __restrict__ whh16_f, const _Float16* __restrict__ whh16_b,
             const _Float16* __restrict__ xproj_f, const _Float16* __restrict__ xproj_b,
             _Float16* __restrict__ hout_f, _Float16* __restrict__ hout_b)
{
  __shared__ __align__(16) _Float16 h_lds[B_ * H_];   // 16 KB

  const int dir = blockIdx.x;
  const _Float16* whh16 = dir ? whh16_b : whh16_f;
  const _Float16* xproj = dir ? xproj_b : xproj_f;
  _Float16*       hout  = dir ? hout_b  : hout_f;

  const int tid  = threadIdx.x;
  const int w    = tid >> 5, lane = tid & 31;
  const int mp   = w >> 3,   q8   = w & 7;
  const int ln   = lane & 15, half = lane >> 4;

  for (int i = tid; i < B_ * H_; i += 512) h_lds[i] = (_Float16)0.0f;

  // One-time load of this wave's W_hh fragments -> register resident.
  v16h bf[4][4];                                      // [gate][k] : 128 VGPRs
  #pragma unroll
  for (int G = 0; G < 4; ++G)
    #pragma unroll
    for (int k = 0; k < 4; ++k) {
      const int n = G * 128 + q8 * 16 + ln;           // W_hh row == output column
      const _Float16* wp = whh16 + (size_t)n * H_ + k * 32 + half * 8;
      bf[G][k] = ldfrag(wp, wp + 16);
    }

  float c_reg[2][8];
  #pragma unroll
  for (int mi = 0; mi < 2; ++mi)
    #pragma unroll
    for (int v = 0; v < 8; ++v) c_reg[mi][v] = 0.0f;

  __syncthreads();

  for (int step = 0; step < T_; ++step) {
    const int t = dir ? (T_ - 1 - step) : step;

    // Prefetch next step's x_proj fragments while this step computes.
    if (step + 1 < T_) {
      const int tn = dir ? (T_ - 2 - step) : (step + 1);
      #pragma unroll
      for (int G = 0; G < 4; ++G)
        #pragma unroll
        for (int mi = 0; mi < 2; ++mi)
          __builtin_prefetch(xproj + xp_off(tn, 2 * mp + mi, G, q8, lane), 0, 3);
    }

    // Init accumulators from hoisted input projection: 8 x b128 + cvt.
    v8f acc[4][2];
    #pragma unroll
    for (int G = 0; G < 4; ++G)
      #pragma unroll
      for (int mi = 0; mi < 2; ++mi) {
        const v8h xv = *(const v8h*)(xproj + xp_off(t, 2 * mp + mi, G, q8, lane));
        #pragma unroll
        for (int v = 0; v < 8; ++v) acc[G][mi][v] = (float)xv[v];
      }

    // g += h @ W_hh^T : A from LDS, B already in registers; 32 WMMAs/wave/step.
    #pragma unroll
    for (int k = 0; k < 4; ++k) {
      const _Float16* hp0 = h_lds + (mp * 32 + ln) * H_ + k * 32 + half * 8;
      const _Float16* hp1 = hp0 + 16 * H_;
      const v16h a0 = ldfrag(hp0, hp0 + 16);
      const v16h a1 = ldfrag(hp1, hp1 + 16);
      #pragma unroll
      for (int G = 0; G < 4; ++G) {
        acc[G][0] = WMMA_F16(a0, bf[G][k], acc[G][0]);
        acc[G][1] = WMMA_F16(a1, bf[G][k], acc[G][1]);
      }
    }
    __syncthreads();  // all h_lds reads complete before this step's writes

    // LSTM cell: gate tiles are lane-aligned -> pure in-register update.
    #pragma unroll
    for (int mi = 0; mi < 2; ++mi)
      #pragma unroll
      for (int v = 0; v < 8; ++v) {
        const float i_ = acc[0][mi][v];
        const float f_ = acc[1][mi][v];
        const float g_ = acc[2][mi][v];
        const float o_ = acc[3][mi][v];
        float c = sigmoidf_(f_) * c_reg[mi][v] + sigmoidf_(i_) * tanhf(g_);
        c_reg[mi][v] = c;
        const float h = sigmoidf_(o_) * tanhf(c);
        const int brow = mp * 32 + mi * 16 + v + 8 * half;
        const int hh   = q8 * 16 + ln;
        h_lds[brow * H_ + hh] = (_Float16)h;
        hout[((size_t)t * B_ + brow) * H_ + hh] = (_Float16)h;
      }

    __syncthreads();  // new h visible before next step's A-fragment loads
  }
}

// ---------------------------------------------------------------------------
// FC: logits[t,b,n] = fc_W[n, 0:128].h_f + fc_W[n, 128:256].h_b + fc_b[n]
// N=8 -> VALU dot products (a WMMA N-tile would be 50% padding waste).
// ---------------------------------------------------------------------------
__global__ void k_fc(const _Float16* __restrict__ hf, const _Float16* __restrict__ hb,
                     const float* __restrict__ fcW, const float* __restrict__ fcb,
                     float* __restrict__ logits)
{
  const int idx = blockIdx.x * blockDim.x + threadIdx.x;  // t*B + b
  if (idx >= T_ * B_) return;
  const _Float16* pf = hf + (size_t)idx * H_;
  const _Float16* pb = hb + (size_t)idx * H_;
  #pragma unroll
  for (int n = 0; n < N_; ++n) {
    const float* wr = fcW + n * 2 * H_;
    float s = fcb[n];
    for (int k = 0; k < H_; ++k) s = fmaf((float)pf[k], wr[k], s);
    for (int k = 0; k < H_; ++k) s = fmaf((float)pb[k], wr[H_ + k], s);
    logits[(size_t)idx * N_ + n] = s;
  }
}

// ---------------------------------------------------------------------------
// Viterbi: one lane per batch row (64 lanes), N=8 max-plus scan in registers,
// backpointers spilled to workspace, then backtrace into the int32 path.
// ---------------------------------------------------------------------------
__global__ void k_viterbi(const float* __restrict__ logits,
                          const float* __restrict__ start_trans,
                          const float* __restrict__ end_trans,
                          const float* __restrict__ transitions,
                          unsigned char* __restrict__ bp,
                          int* __restrict__ path)
{
  const int b = threadIdx.x;
  __shared__ float tr[N_ * N_];
  if (b < N_ * N_) tr[b] = transitions[b];
  __syncthreads();

  float sc[N_], ns[N_];
  #pragma unroll
  for (int n = 0; n < N_; ++n) sc[n] = start_trans[n] + logits[(size_t)b * N_ + n];

  for (int t = 1; t < T_; ++t) {
    const float* em = logits + ((size_t)t * B_ + b) * N_;
    #pragma unroll
    for (int j = 0; j < N_; ++j) {
      float best = sc[0] + tr[j];
      int arg = 0;
      #pragma unroll
      for (int i = 1; i < N_; ++i) {
        const float v = sc[i] + tr[i * N_ + j];
        if (v > best) { best = v; arg = i; }   // '>' keeps first argmax like jnp
      }
      ns[j] = best + em[j];
      bp[((size_t)(t - 1) * B_ + b) * N_ + j] = (unsigned char)arg;
    }
    #pragma unroll
    for (int j = 0; j < N_; ++j) sc[j] = ns[j];
  }

  int last = 0;
  float best = sc[0] + end_trans[0];
  #pragma unroll
  for (int n = 1; n < N_; ++n) {
    const float v = sc[n] + end_trans[n];
    if (v > best) { best = v; last = n; }
  }
  path[b * T_ + (T_ - 1)] = last;
  for (int t = T_ - 2; t >= 0; --t) {
    last = bp[((size_t)t * B_ + b) * N_ + last];
    path[b * T_ + t] = last;
  }
}

// ---------------------------------------------------------------------------
extern "C" void kernel_launch(void* const* d_in, const int* in_sizes, int n_in,
                              void* d_out, int out_size, void* d_ws, size_t ws_size,
                              hipStream_t stream)
{
  const int*   x         = (const int*)  d_in[0];
  const float* embedding = (const float*)d_in[1];
  const float* W_ih_f    = (const float*)d_in[2];
  const float* W_hh_f    = (const float*)d_in[3];
  const float* b_f       = (const float*)d_in[4];
  const float* W_ih_b    = (const float*)d_in[5];
  const float* W_hh_b    = (const float*)d_in[6];
  const float* b_b       = (const float*)d_in[7];
  const float* fc_W      = (const float*)d_in[8];
  const float* fc_b      = (const float*)d_in[9];
  const float* start_tr  = (const float*)d_in[10];
  const float* end_tr    = (const float*)d_in[11];
  const float* trans     = (const float*)d_in[12];

  char* ws = (char*)d_ws;
  size_t o = 0;
  _Float16* xproj_f = (_Float16*)(ws + o); o += (size_t)T_ * B_ * G4H * 2;  // 64 MB
  _Float16* xproj_b = (_Float16*)(ws + o); o += (size_t)T_ * B_ * G4H * 2;  // 64 MB
  _Float16* hout_f  = (_Float16*)(ws + o); o += (size_t)T_ * B_ * H_ * 2;   // 16 MB
  _Float16* hout_b  = (_Float16*)(ws + o); o += (size_t)T_ * B_ * H_ * 2;   // 16 MB
  _Float16* wih16_f = (_Float16*)(ws + o); o += (size_t)G4H * E_ * 2;
  _Float16* wih16_b = (_Float16*)(ws + o); o += (size_t)G4H * E_ * 2;
  _Float16* whh16_f = (_Float16*)(ws + o); o += (size_t)G4H * H_ * 2;
  _Float16* whh16_b = (_Float16*)(ws + o); o += (size_t)G4H * H_ * 2;
  _Float16* emb16   = (_Float16*)(ws + o); o += (size_t)V_ * E_ * 2;
  float*    logits  = (float*)   (ws + o); o += (size_t)T_ * B_ * N_ * 4;   // 2 MB
  unsigned char* bp = (unsigned char*)(ws + o); o += (size_t)(T_ - 1) * B_ * N_;

  // 1) weights/embedding -> f16 (L2-resident thereafter)
  k_f32_to_f16<<<(G4H * E_ + 255) / 256, 256, 0, stream>>>(W_ih_f, wih16_f, G4H * E_);
  k_f32_to_f16<<<(G4H * E_ + 255) / 256, 256, 0, stream>>>(W_ih_b, wih16_b, G4H * E_);
  k_f32_to_f16<<<(G4H * H_ + 255) / 256, 256, 0, stream>>>(W_hh_f, whh16_f, G4H * H_);
  k_f32_to_f16<<<(G4H * H_ + 255) / 256, 256, 0, stream>>>(W_hh_b, whh16_b, G4H * H_);
  k_f32_to_f16<<<(V_ * E_ + 255) / 256, 256, 0, stream>>>(embedding, emb16, V_ * E_);

  // 2) hoisted input GEMM (WMMA), both directions, permuted-fragment output
  k_input_proj<<<dim3(T_, 2), 512, 0, stream>>>(x, emb16, wih16_f, wih16_b,
                                                b_f, b_b, xproj_f, xproj_b);

  // 3) persistent recurrence (WMMA, register-resident W_hh), one block/direction
  k_lstm_recur<<<2, 512, 0, stream>>>(whh16_f, whh16_b, xproj_f, xproj_b,
                                      hout_f, hout_b);

  // 4) FC -> emissions
  k_fc<<<(T_ * B_ + 255) / 256, 256, 0, stream>>>(hout_f, hout_b, fc_W, fc_b, logits);

  // 5) Viterbi decode -> int32 path [B, T]
  k_viterbi<<<1, B_, 0, stream>>>(logits, start_tr, end_tr, trans, bp, (int*)d_out);

  (void)in_sizes; (void)n_in; (void)out_size; (void)ws_size;
}
